// LocalMultiheadSelfAttention_65420941852851
// MI455X (gfx1250) — matrix operand order repaired and checked by reading, not compiled, not run on hardware
//
#include <hip/hip_runtime.h>
#include <hip/hip_bf16.h>

#define L_SEQ   2048
#define D_MODEL 768
#define NHEAD   12
#define HEADD   64
#define N3      (3 * D_MODEL)
#define WIN     128

typedef __attribute__((ext_vector_type(16))) _Float16 v16h;
typedef __attribute__((ext_vector_type(8)))  float    v8f;
typedef int v4i __attribute__((vector_size(16)));

// gfx1250 async global->LDS path (ASYNCcnt), with portable fallback.
#if defined(__gfx1250__) && __has_builtin(__builtin_amdgcn_global_load_async_to_lds_b128) && __has_builtin(__builtin_amdgcn_s_wait_asynccnt)
#define ATHENA_ASYNC_LDS 1
#else
#define ATHENA_ASYNC_LDS 0
#endif

#if ATHENA_ASYNC_LDS
typedef __attribute__((address_space(1))) v4i gas_v4i;   // global
typedef __attribute__((address_space(3))) v4i las_v4i;   // LDS
__device__ __forceinline__ void async_cp16(const _Float16* g, _Float16* l) {
    __builtin_amdgcn_global_load_async_to_lds_b128(
        (gas_v4i*)g, (las_v4i*)l, 0, 0);
}
#endif

__device__ __forceinline__ void sync_cp16(const _Float16* g, _Float16* l) {
    *(float4*)l = *(const float4*)g;
}

// ---------------------------------------------------------------------------
// Conversions
// ---------------------------------------------------------------------------
__global__ void cvt_f32_f16_kernel(const float* __restrict__ in,
                                   _Float16* __restrict__ out, int n) {
    int i = blockIdx.x * blockDim.x + threadIdx.x;
    if (i < n) out[i] = (_Float16)in[i];
}

// in: [rows][cols] f32  ->  out: [cols][rows] f16   (weight transpose for B-frags)
__global__ void transpose_cvt_kernel(const float* __restrict__ in,
                                     _Float16* __restrict__ out,
                                     int rows, int cols) {
    int i = blockIdx.x * blockDim.x + threadIdx.x;
    if (i < rows * cols) {
        int r = i / cols, c = i - r * cols;
        out[(size_t)c * rows + r] = (_Float16)in[i];
    }
}

// ---------------------------------------------------------------------------
// GEMM: C[M][N] = A[M][K](f16) * Bt[N][K]^T(f16) + bias[N], f32 accumulate.
// 256 threads / 8 waves per block; block owns a 128x64 C macro-tile.
// Per K-step(32): A-tile(128x32) + B-tile(64x32) staged in LDS, double
// buffered via async-to-LDS DMA overlapped with the previous step's WMMAs.
// Fragment layouts (wave32, v_wmma_f32_16x16x32_f16):
//   A: lane m=lane&15, g=lane>>4; e:    K=16*g+e  (contiguous, same order as B)
//   B: lane n=lane&15, g=lane>>4; e:    K=16*g+e  (from Bt[N][K], contiguous)
//   C: lane n=lane&15, g=lane>>4; vgpr r -> M=r+8g
// ---------------------------------------------------------------------------
#define LDA_SH 40   // padded LDS row stride (halfs): 80B = 16B-aligned, 16 distinct banks

__global__ void __launch_bounds__(256)
wmma_gemm_f16_kernel(const _Float16* __restrict__ A,
                     const _Float16* __restrict__ Bt,
                     const float* __restrict__ bias,
                     float* __restrict__ C,
                     int M, int N, int K) {
    __shared__ _Float16 Abuf[2][128 * LDA_SH];
    __shared__ _Float16 Bbuf[2][64 * LDA_SH];

    const int tid  = threadIdx.x;
    const int lane = tid & 31;
    const int wave = tid >> 5;

    const int nT = N >> 6;
    const int m0 = (blockIdx.x / nT) * 128;
    const int n0 = (blockIdx.x % nT) * 64;
    if (m0 >= M) return;

    // 16B-chunk coords: A has 128 rows x 4 chunks = 512 (2/thread), B 64x4 = 256 (1/thread)
    const int ar0 = tid >> 2, ar1 = (tid + 256) >> 2, ac = (tid & 3) * 8;
    const int br  = tid >> 2;

    auto stage = [&](int k0, int buf) {
        const _Float16* ga0 = A  + (size_t)(m0 + ar0) * K + k0 + ac;
        const _Float16* ga1 = A  + (size_t)(m0 + ar1) * K + k0 + ac;
        const _Float16* gb  = Bt + (size_t)(n0 + br)  * K + k0 + ac;
        _Float16* la0 = &Abuf[buf][ar0 * LDA_SH + ac];
        _Float16* la1 = &Abuf[buf][ar1 * LDA_SH + ac];
        _Float16* lb  = &Bbuf[buf][br  * LDA_SH + ac];
#if ATHENA_ASYNC_LDS
        async_cp16(ga0, la0);
        async_cp16(ga1, la1);
        async_cp16(gb,  lb);
#else
        sync_cp16(ga0, la0);
        sync_cp16(ga1, la1);
        sync_cp16(gb,  lb);
#endif
    };

    v8f acc[4] = {};
    const int steps = K >> 5;

    stage(0, 0);
    for (int i = 0; i < steps; ++i) {
#if ATHENA_ASYNC_LDS
        __builtin_amdgcn_s_wait_asynccnt(0);   // buf[i&1] landed in LDS
#endif
        __syncthreads();
        if (i + 1 < steps) stage((i + 1) << 5, (i + 1) & 1);  // DMA overlaps WMMAs below

        const _Float16* Ab = Abuf[i & 1];
        const _Float16* Bb = Bbuf[i & 1];
        const int g = lane >> 4;

        // Hoist ALL fragment loads ahead of the WMMA group so the compiler can
        // clause the ds_load_b128s and issue the 4 WMMAs back-to-back.
        v16h a;
        {
            const _Float16* p = Ab + (wave * 16 + (lane & 15)) * LDA_SH + g * 16;
#pragma unroll
            for (int e = 0; e < 8; ++e) a[e] = p[e];
#pragma unroll
            for (int e = 0; e < 8; ++e) a[8 + e] = p[8 + e];
        }
        v16h b[4];
#pragma unroll
        for (int t = 0; t < 4; ++t) {
            const _Float16* p = Bb + (t * 16 + (lane & 15)) * LDA_SH + g * 16;
#pragma unroll
            for (int e = 0; e < 16; ++e) b[t][e] = p[e];
        }
#pragma unroll
        for (int t = 0; t < 4; ++t) {
            acc[t] = __builtin_amdgcn_wmma_f32_16x16x32_f16(
                false, a, false, b[t], (short)0, acc[t], false, false);
        }
        __syncthreads();
    }

    const int g  = lane >> 4;
    const int nn = lane & 15;
    const int mw = m0 + wave * 16;
#pragma unroll
    for (int t = 0; t < 4; ++t) {
        const int n = n0 + 16 * t + nn;
        const float bv = bias[n];
#pragma unroll
        for (int r = 0; r < 8; ++r)
            C[(size_t)(mw + r + 8 * g) * N + n] = acc[t][r] + bv;
    }
}

// ---------------------------------------------------------------------------
// RoPE (half-split, d=64) + split qkv[L][2304] -> q/k/v [H][L][64] f32
// ---------------------------------------------------------------------------
__global__ void rope_split_kernel(const float* __restrict__ qkv,
                                  float* __restrict__ q,
                                  float* __restrict__ k,
                                  float* __restrict__ v, int L) {
    const int idx = blockIdx.x * blockDim.x + threadIdx.x;
    if (idx >= NHEAD * L * 32) return;
    const int i = idx & 31;
    const int l = (idx >> 5) % L;
    const int h = idx / (32 * L);

    const float* row = qkv + (size_t)l * N3;
    const float ang = (float)l * powf(10000.0f, -(float)(2 * i) / 64.0f);
    const float c = cosf(ang), s = sinf(ang);
    const size_t ob = ((size_t)h * L + l) * HEADD;

    float x1 = row[h * HEADD + i], x2 = row[h * HEADD + i + 32];
    q[ob + i]      = x1 * c - x2 * s;
    q[ob + i + 32] = x1 * s + x2 * c;

    x1 = row[D_MODEL + h * HEADD + i];
    x2 = row[D_MODEL + h * HEADD + i + 32];
    k[ob + i]      = x1 * c - x2 * s;
    k[ob + i + 32] = x1 * s + x2 * c;

    v[ob + i]      = row[2 * D_MODEL + h * HEADD + i];
    v[ob + i + 32] = row[2 * D_MODEL + h * HEADD + i + 32];
}

// ---------------------------------------------------------------------------
// Windowed attention: one block of 128 threads per (h, l).
// ---------------------------------------------------------------------------
__global__ void window_attn_kernel(const float* __restrict__ q,
                                   const float* __restrict__ k,
                                   const float* __restrict__ v,
                                   const int* __restrict__ mask,
                                   float* __restrict__ out, int L) {
    __shared__ float qsh[HEADD];
    __shared__ float sc[WIN], pw[WIN], tmp[WIN];
    __shared__ int   sidx[WIN];

    const int h = blockIdx.x / L;
    const int l = blockIdx.x - h * L;
    const int j = threadIdx.x;
    const float scale = rsqrtf((float)D_MODEL);   // reference scales by 1/sqrt(768)

    if (j < HEADD) qsh[j] = q[((size_t)h * L + l) * HEADD + j];
    __syncthreads();

    const int raw = l + j - WIN / 2;
    int idx = raw < 0 ? 0 : (raw >= L ? L - 1 : raw);
    const bool valid = (raw >= 0) && (raw < L) && (mask[idx] != 0);
    const float* kr = k + ((size_t)h * L + idx) * HEADD;
    float s = 0.f;
#pragma unroll
    for (int d = 0; d < HEADD; ++d) s += qsh[d] * kr[d];
    sc[j]   = valid ? s * scale : -1e30f;
    sidx[j] = idx;
    tmp[j]  = sc[j];
    __syncthreads();

    for (int st = WIN / 2; st > 0; st >>= 1) {
        if (j < st) tmp[j] = fmaxf(tmp[j], tmp[j + st]);
        __syncthreads();
    }
    const float mx = tmp[0];
    __syncthreads();

    const float e = expf(sc[j] - mx);
    pw[j]  = e;
    tmp[j] = e;
    __syncthreads();
    for (int st = WIN / 2; st > 0; st >>= 1) {
        if (j < st) tmp[j] += tmp[j + st];
        __syncthreads();
    }
    const float sum = tmp[0];
    __syncthreads();

    if (j < HEADD) {
        float acc = 0.f;
        for (int jj = 0; jj < WIN; ++jj)
            acc += pw[jj] * v[((size_t)h * L + sidx[jj]) * HEADD + j];
        out[(size_t)l * D_MODEL + h * HEADD + j] = acc / sum;
    }
}

// ---------------------------------------------------------------------------
// CLS row: full attention of q[h][0] over all L keys; overwrites out row 0.
// ---------------------------------------------------------------------------
__global__ void cls_attn_kernel(const float* __restrict__ q,
                                const float* __restrict__ k,
                                const float* __restrict__ v,
                                const int* __restrict__ mask,
                                float* __restrict__ out, int L) {
    __shared__ float p[L_SEQ];
    __shared__ float red[256];
    __shared__ float qsh[HEADD];
    const int h = blockIdx.x;
    const int t = threadIdx.x;
    const float scale = rsqrtf((float)D_MODEL);

    if (t < HEADD) qsh[t] = q[(size_t)h * L * HEADD + t];
    __syncthreads();

    float lmax = -1e30f;
    for (int jj = t; jj < L; jj += 256) {
        const float* kr = k + ((size_t)h * L + jj) * HEADD;
        __builtin_prefetch(kr + 256 * HEADD, 0, 0);   // -> global_prefetch_b8
        float s = 0.f;
#pragma unroll
        for (int d = 0; d < HEADD; ++d) s += qsh[d] * kr[d];
        s = (mask[jj] != 0) ? s * scale : -1e30f;
        p[jj] = s;
        lmax = fmaxf(lmax, s);
    }
    red[t] = lmax;
    __syncthreads();
    for (int st = 128; st > 0; st >>= 1) {
        if (t < st) red[t] = fmaxf(red[t], red[t + st]);
        __syncthreads();
    }
    const float mx = red[0];
    __syncthreads();

    float lsum = 0.f;
    for (int jj = t; jj < L; jj += 256) {
        const float e = expf(p[jj] - mx);
        p[jj] = e;
        lsum += e;
    }
    red[t] = lsum;
    __syncthreads();
    for (int st = 128; st > 0; st >>= 1) {
        if (t < st) red[t] += red[t + st];
        __syncthreads();
    }
    const float sum = red[0];
    __syncthreads();

    if (t < HEADD) {
        float acc = 0.f;
        for (int jj = 0; jj < L; ++jj)
            acc += p[jj] * v[((size_t)h * L + jj) * HEADD + t];
        out[(size_t)h * HEADD + t] = acc / sum;   // row l = 0
    }
}

// ---------------------------------------------------------------------------
// Launcher
// ---------------------------------------------------------------------------
extern "C" void kernel_launch(void* const* d_in, const int* in_sizes, int n_in,
                              void* d_out, int out_size, void* d_ws, size_t ws_size,
                              hipStream_t stream) {
    (void)in_sizes; (void)n_in; (void)out_size; (void)ws_size;
    const int L = L_SEQ, D = D_MODEL, H = NHEAD;

    const float* emb  = (const float*)d_in[0];
    const int*   mask = (const int*)d_in[1];
    const float* Wqkv = (const float*)d_in[2];
    const float* bqkv = (const float*)d_in[3];
    const float* Wo   = (const float*)d_in[4];
    const float* bo   = (const float*)d_in[5];
    float* out = (float*)d_out;

    char* w = (char*)d_ws;
    size_t off = 0;
    auto carve = [&](size_t bytes) -> void* {
        void* p = (void*)(w + off);
        off = (off + bytes + 255) & ~(size_t)255;
        return p;
    };
    _Float16* X16   = (_Float16*)carve((size_t)L * D * sizeof(_Float16));
    _Float16* WqkvT = (_Float16*)carve((size_t)N3 * D * sizeof(_Float16));
    _Float16* WoT   = (_Float16*)carve((size_t)D * D * sizeof(_Float16));
    float*    qkv   = (float*)carve((size_t)L * N3 * sizeof(float));
    float*    qbuf  = (float*)carve((size_t)H * L * HEADD * sizeof(float));
    float*    kbuf  = (float*)carve((size_t)H * L * HEADD * sizeof(float));
    float*    vbuf  = (float*)carve((size_t)H * L * HEADD * sizeof(float));
    float*    attn  = (float*)carve((size_t)L * D * sizeof(float));
    _Float16* A16   = (_Float16*)carve((size_t)L * D * sizeof(_Float16));

    const int nXD = L * D;
    cvt_f32_f16_kernel<<<(nXD + 255) / 256, 256, 0, stream>>>(emb, X16, nXD);
    const int nW1 = D * N3;
    transpose_cvt_kernel<<<(nW1 + 255) / 256, 256, 0, stream>>>(Wqkv, WqkvT, D, N3);
    const int nW2 = D * D;
    transpose_cvt_kernel<<<(nW2 + 255) / 256, 256, 0, stream>>>(Wo, WoT, D, D);

    // QKV projection: M=2048, N=2304, K=768 -> (2048/128)*(2304/64) = 576 blocks
    wmma_gemm_f16_kernel<<<(L / 128) * (N3 / 64), 256, 0, stream>>>(
        X16, WqkvT, bqkv, qkv, L, N3, D);

    const int nRP = H * L * 32;
    rope_split_kernel<<<(nRP + 255) / 256, 256, 0, stream>>>(qkv, qbuf, kbuf, vbuf, L);

    window_attn_kernel<<<H * L, WIN, 0, stream>>>(qbuf, kbuf, vbuf, mask, attn, L);
    cls_attn_kernel<<<H, 256, 0, stream>>>(qbuf, kbuf, vbuf, mask, attn, L);

    cvt_f32_f16_kernel<<<(nXD + 255) / 256, 256, 0, stream>>>(attn, A16, nXD);

    // Output projection: M=2048, N=768, K=768 -> (2048/128)*(768/64) = 192 blocks
    wmma_gemm_f16_kernel<<<(L / 128) * (D / 64), 256, 0, stream>>>(
        A16, WoT, bo, out, L, D, D);
}
